// ImprovedLSTMCell_49306224558689
// MI455X (gfx1250) — matrix operand order repaired and checked by reading, not compiled, not run on hardware
//
#include <hip/hip_runtime.h>

typedef __attribute__((ext_vector_type(16))) __bf16 v16bf;
typedef __attribute__((ext_vector_type(8)))  float  v8f;

#define Bsz 16384
#define Dd  1024
#define Hh  1024

#define BM 128      // batch rows per block
#define BN 32       // h columns per block (x5 gates internally)
#define KT 32       // K per WMMA step
#define PAD 8       // LDS row pad (shorts) -> 80B row stride
#define NSTEP (Dd / KT)   // 32 K-steps per phase

// bf16 workspace layout (elements)
#define XB_ELEMS (Bsz * Dd)
#define HB_ELEMS (Bsz * Hh)
#define WB_ELEMS (5 * Hh * Dd)
#define WS_NEED_BYTES ((size_t)(XB_ELEMS + HB_ELEMS + 2 * WB_ELEMS) * 2)

union FragAB { v16bf v; uint4 q[2]; };

__device__ __forceinline__ unsigned pack_bf16(float a, float b) {
  unsigned ua = __float_as_uint(a);
  unsigned ub = __float_as_uint(b);
  ua += 0x7FFFu + ((ua >> 16) & 1u);
  ub += 0x7FFFu + ((ub >> 16) & 1u);
  return (ua >> 16) | (ub & 0xFFFF0000u);
}
__device__ __forceinline__ float sigm(float x) { return 1.0f / (1.0f + __expf(-x)); }
__device__ __forceinline__ float tanh_fast(float x) { return 1.0f - 2.0f / (__expf(2.0f * x) + 1.0f); }
__device__ __forceinline__ unsigned lds_off(const void* p) {
  return (unsigned)(uintptr_t)p;   // generic LDS addr low 32 bits == LDS byte offset
}

// ---------------- pass 1: fp32 -> bf16 (RNE), pure bandwidth ----------------
__global__ __launch_bounds__(256) void cvt_bf16(const float4* __restrict__ src,
                                                uint2* __restrict__ dst, int n4) {
  int i = blockIdx.x * blockDim.x + threadIdx.x;
  const int stride = gridDim.x * blockDim.x;
  for (; i < n4; i += stride) {
    const float4 f = src[i];
    uint2 p;
    p.x = pack_bf16(f.x, f.y);
    p.y = pack_bf16(f.z, f.w);
    dst[i] = p;
  }
}

// ============= main kernel: bf16 sources + async global->LDS DMA =============
__global__ __launch_bounds__(256) void lstm_wmma_async(
    const unsigned short* __restrict__ xb, const unsigned short* __restrict__ hb,
    const unsigned short* __restrict__ wxb, const unsigned short* __restrict__ uhb,
    const float* __restrict__ c_prev, const float* __restrict__ dw,
    const float* __restrict__ bx, const float* __restrict__ bh,
    float* __restrict__ out)
{
  __shared__ unsigned short sA[2][BM][KT + PAD];       // 2 x 10240 B
  __shared__ unsigned short sB[2][5 * BN][KT + PAD];   // 2 x 12800 B

  const int tid  = threadIdx.x;
  const int lane = tid & 31;
  const int wave = tid >> 5;
  const int rowBase = blockIdx.y * BM;
  const int hBase   = blockIdx.x * BN;

  v8f acc[5][2] = {};

  // ---- async-copy maps: A = 2x B128 chunks/thread, B = 5x B64 chunks/thread ----
  unsigned aLds[2]; size_t aGOff[2];          // byte offsets
  #pragma unroll
  for (int t = 0; t < 2; ++t) {
    const int i = tid + 256 * t;              // 16B-chunk index 0..511
    const int arow = i >> 2, ac16 = i & 3;
    aLds[t]  = lds_off(&sA[0][arow][ac16 * 8]);
    aGOff[t] = ((size_t)(rowBase + arow) * Dd + ac16 * 8) * 2;
  }
  unsigned bLds[5]; size_t bGOff[5];
  #pragma unroll
  for (int t = 0; t < 5; ++t) {
    const int i = tid + 256 * t;              // 8B-chunk index 0..1279
    const int brow = i >> 3, bc8 = i & 7;
    const int g = brow >> 5, n = brow & 31;
    bLds[t]  = lds_off(&sB[0][brow][bc8 * 4]);
    bGOff[t] = ((size_t)(g * Hh + hBase + n) * Dd + bc8 * 4) * 2;
  }
  const unsigned bufStrideA = (unsigned)sizeof(sA[0]);
  const unsigned bufStrideB = (unsigned)sizeof(sB[0]);

  // fragment maps (wave32 WMMA 16-bit layouts)
  const int fm  = lane & 15;
  const int akh = (lane >> 4) * 8;
  const int bkh = (lane >> 4) * 8;

  auto issueTile = [&](const unsigned short* __restrict__ Ap,
                       const unsigned short* __restrict__ Wp, int buf, int k0) {
    const size_t kByte = (size_t)k0 * 2;
    #pragma unroll
    for (int t = 0; t < 2; ++t) {
      const unsigned ld = aLds[t] + buf * bufStrideA;
      const unsigned long long ga =
          (unsigned long long)(uintptr_t)((const char*)Ap + aGOff[t] + kByte);
      asm volatile("global_load_async_to_lds_b128 %0, %1, off"
                   :: "v"(ld), "v"(ga) : "memory");
    }
    #pragma unroll
    for (int t = 0; t < 5; ++t) {
      const unsigned ld = bLds[t] + buf * bufStrideB;
      const unsigned long long ga =
          (unsigned long long)(uintptr_t)((const char*)Wp + bGOff[t] + kByte);
      asm volatile("global_load_async_to_lds_b64 %0, %1, off"
                   :: "v"(ld), "v"(ga) : "memory");
    }
  };

  auto computeTile = [&](int buf) {
    FragAB a;
    a.q[0] = *reinterpret_cast<const uint4*>(&sA[buf][wave * 16 + fm][akh]);
    a.q[1] = *reinterpret_cast<const uint4*>(&sA[buf][wave * 16 + fm][16 + akh]);
    #pragma unroll
    for (int g = 0; g < 5; ++g) {
      #pragma unroll
      for (int j = 0; j < 2; ++j) {
        FragAB b;
        const unsigned short* rowp = &sB[buf][g * 32 + j * 16 + fm][bkh];
        b.q[0] = *reinterpret_cast<const uint4*>(rowp);
        b.q[1] = *reinterpret_cast<const uint4*>(rowp + 16);
        acc[g][j] = __builtin_amdgcn_wmma_f32_16x16x32_bf16(
            false, a.v, false, b.v, (short)0, acc[g][j], false, false);
      }
    }
  };

  auto phase = [&](const unsigned short* __restrict__ Ap,
                   const unsigned short* __restrict__ Wp) {
    issueTile(Ap, Wp, 0, 0);
    asm volatile("s_wait_asynccnt 0x0" ::: "memory");
    __syncthreads();
    for (int s = 0; s < NSTEP; ++s) {                     // uniform control flow
      if (s + 1 < NSTEP) issueTile(Ap, Wp, (s + 1) & 1, (s + 1) * KT);
      computeTile(s & 1);                                 // ds_load + 10x WMMA
      if (s + 1 < NSTEP) asm volatile("s_wait_asynccnt 0x0" ::: "memory");
      __syncthreads();
    }
  };

  phase(xb, wxb);   // gates += x · Wx^T
  phase(hb, uhb);   // gates += h · Uh^T

  // ---- fused LSTM epilogue (fp32) ----
  const int strip = rowBase + wave * 16;
  const int mofs  = (lane < 16) ? 0 : 8;
  const int nlane = lane & 15;
  #pragma unroll
  for (int j = 0; j < 2; ++j) {
    const int col = hBase + j * 16 + nlane;
    float bias[5];
    #pragma unroll
    for (int g = 0; g < 5; ++g) bias[g] = bx[g * Hh + col] + bh[g * Hh + col];
    #pragma unroll
    for (int r = 0; r < 8; ++r) {
      const int idx = (strip + mofs + r) * Hh + col;
      const float dwv = dw[idx];
      const float cpv = c_prev[idx];
      const float gi = acc[0][j][r] + bias[0];
      const float gf = acc[1][j][r] + bias[1];
      const float go = acc[2][j][r] + bias[2];
      const float gc = acc[3][j][r] + bias[3];
      const float gs = acc[4][j][r] + bias[4];
      const float i_t = sigm(gi), f_t = sigm(gf), o_t = sigm(go);
      const float ch  = tanh_fast(gc);
      const float s_t = sigm(gs) * dwv;
      const float c_t = f_t * cpv + i_t * ch * s_t;
      const float h_t = o_t * tanh_fast(c_t);
      out[idx] = h_t;
      out[Bsz * Hh + idx] = c_t;
    }
  }
}

// ============= fallback: fused kernel with fp32 loads + in-reg packing =============
__global__ __launch_bounds__(256) void lstm_wmma_fp32src(
    const float* __restrict__ x,  const float* __restrict__ h_prev,
    const float* __restrict__ c_prev, const float* __restrict__ dw,
    const float* __restrict__ Wx, const float* __restrict__ bx,
    const float* __restrict__ Uh, const float* __restrict__ bh,
    float* __restrict__ out)
{
  __shared__ unsigned short sA[2][BM][KT + PAD];
  __shared__ unsigned short sB[2][5 * BN][KT + PAD];

  const int tid  = threadIdx.x;
  const int lane = tid & 31;
  const int wave = tid >> 5;
  const int rowBase = blockIdx.y * BM;
  const int hBase   = blockIdx.x * BN;

  v8f acc[5][2] = {};

  const int ar = tid >> 1;
  const int ac = (tid & 1) * 16;
  const size_t aOff = (size_t)(rowBase + ar) * Dd + ac;

  int bGr[5], bC4[5]; size_t bOff[5];
  #pragma unroll
  for (int t = 0; t < 5; ++t) {
    const int i = tid + 256 * t;
    const int gr = i >> 3, c4 = i & 7;
    const int g = gr >> 5, n = gr & 31;
    bGr[t] = gr; bC4[t] = c4;
    bOff[t] = (size_t)(g * Hh + hBase + n) * Dd + c4 * 4;
  }

  const int fm  = lane & 15;
  const int akh = (lane >> 4) * 8;
  const int bkh = (lane >> 4) * 8;

  float4 ra[4]; float4 rb[5];

  auto loadTile = [&](const float* __restrict__ Ap, const float* __restrict__ Wp, int k0) {
    const float4* ap = reinterpret_cast<const float4*>(Ap + aOff + k0);
    ra[0] = ap[0]; ra[1] = ap[1]; ra[2] = ap[2]; ra[3] = ap[3];
    #pragma unroll
    for (int t = 0; t < 5; ++t)
      rb[t] = *reinterpret_cast<const float4*>(Wp + bOff[t] + k0);
  };
  auto storeTile = [&](int buf) {
    uint4 p0, p1;
    p0.x = pack_bf16(ra[0].x, ra[0].y); p0.y = pack_bf16(ra[0].z, ra[0].w);
    p0.z = pack_bf16(ra[1].x, ra[1].y); p0.w = pack_bf16(ra[1].z, ra[1].w);
    p1.x = pack_bf16(ra[2].x, ra[2].y); p1.y = pack_bf16(ra[2].z, ra[2].w);
    p1.z = pack_bf16(ra[3].x, ra[3].y); p1.w = pack_bf16(ra[3].z, ra[3].w);
    uint4* dst = reinterpret_cast<uint4*>(&sA[buf][ar][ac]);
    dst[0] = p0; dst[1] = p1;
    #pragma unroll
    for (int t = 0; t < 5; ++t) {
      uint2 p;
      p.x = pack_bf16(rb[t].x, rb[t].y);
      p.y = pack_bf16(rb[t].z, rb[t].w);
      *reinterpret_cast<uint2*>(&sB[buf][bGr[t]][bC4[t] * 4]) = p;
    }
  };
  auto computeTile = [&](int buf) {
    FragAB a;
    a.q[0] = *reinterpret_cast<const uint4*>(&sA[buf][wave * 16 + fm][akh]);
    a.q[1] = *reinterpret_cast<const uint4*>(&sA[buf][wave * 16 + fm][16 + akh]);
    #pragma unroll
    for (int g = 0; g < 5; ++g) {
      #pragma unroll
      for (int j = 0; j < 2; ++j) {
        FragAB b;
        const unsigned short* rowp = &sB[buf][g * 32 + j * 16 + fm][bkh];
        b.q[0] = *reinterpret_cast<const uint4*>(rowp);
        b.q[1] = *reinterpret_cast<const uint4*>(rowp + 16);
        acc[g][j] = __builtin_amdgcn_wmma_f32_16x16x32_bf16(
            false, a.v, false, b.v, (short)0, acc[g][j], false, false);
      }
    }
  };
  auto phase = [&](const float* __restrict__ Ap, const float* __restrict__ Wp) {
    loadTile(Ap, Wp, 0);
    storeTile(0);
    __syncthreads();
    for (int s = 0; s < NSTEP; ++s) {
      if (s + 1 < NSTEP) loadTile(Ap, Wp, (s + 1) * KT);
      computeTile(s & 1);
      if (s + 1 < NSTEP) storeTile((s + 1) & 1);
      __syncthreads();
    }
  };

  phase(x, Wx);
  phase(h_prev, Uh);

  const int strip = rowBase + wave * 16;
  const int mofs  = (lane < 16) ? 0 : 8;
  const int nlane = lane & 15;
  #pragma unroll
  for (int j = 0; j < 2; ++j) {
    const int col = hBase + j * 16 + nlane;
    float bias[5];
    #pragma unroll
    for (int g = 0; g < 5; ++g) bias[g] = bx[g * Hh + col] + bh[g * Hh + col];
    #pragma unroll
    for (int r = 0; r < 8; ++r) {
      const int idx = (strip + mofs + r) * Hh + col;
      const float dwv = dw[idx];
      const float cpv = c_prev[idx];
      const float gi = acc[0][j][r] + bias[0];
      const float gf = acc[1][j][r] + bias[1];
      const float go = acc[2][j][r] + bias[2];
      const float gc = acc[3][j][r] + bias[3];
      const float gs = acc[4][j][r] + bias[4];
      const float i_t = sigm(gi), f_t = sigm(gf), o_t = sigm(go);
      const float ch  = tanh_fast(gc);
      const float s_t = sigm(gs) * dwv;
      const float c_t = f_t * cpv + i_t * ch * s_t;
      const float h_t = o_t * tanh_fast(c_t);
      out[idx] = h_t;
      out[Bsz * Hh + idx] = c_t;
    }
  }
}

extern "C" void kernel_launch(void* const* d_in, const int* in_sizes, int n_in,
                              void* d_out, int out_size, void* d_ws, size_t ws_size,
                              hipStream_t stream) {
  const float* x  = (const float*)d_in[0];
  const float* h  = (const float*)d_in[1];
  const float* c  = (const float*)d_in[2];
  const float* dw = (const float*)d_in[3];
  const float* Wx = (const float*)d_in[4];
  const float* bx = (const float*)d_in[5];
  const float* Uh = (const float*)d_in[6];
  const float* bh = (const float*)d_in[7];
  float* out = (float*)d_out;

  dim3 grid(Hh / BN, Bsz / BM);   // (32, 128)

  if (ws_size >= WS_NEED_BYTES) {
    unsigned short* xb  = (unsigned short*)d_ws;
    unsigned short* hb  = xb + XB_ELEMS;
    unsigned short* wxb = hb + HB_ELEMS;
    unsigned short* uhb = wxb + WB_ELEMS;
    // pass 1: fp32 -> bf16 mirrors (bandwidth-bound)
    cvt_bf16<<<2048, 256, 0, stream>>>((const float4*)x,  (uint2*)xb,  XB_ELEMS / 4);
    cvt_bf16<<<2048, 256, 0, stream>>>((const float4*)h,  (uint2*)hb,  HB_ELEMS / 4);
    cvt_bf16<<<1024, 256, 0, stream>>>((const float4*)Wx, (uint2*)wxb, WB_ELEMS / 4);
    cvt_bf16<<<1024, 256, 0, stream>>>((const float4*)Uh, (uint2*)uhb, WB_ELEMS / 4);
    // pass 2: WMMA GEMM with async global->LDS staging
    lstm_wmma_async<<<grid, 256, 0, stream>>>(xb, hb, wxb, uhb, c, dw, bx, bh, out);
  } else {
    lstm_wmma_fp32src<<<grid, 256, 0, stream>>>(x, h, c, dw, Wx, bx, Uh, bh, out);
  }
}
